// DinoSAM_OCextractor_30176440222228
// MI455X (gfx1250) — matrix operand order repaired and checked by reading, not compiled, not run on hardware
//
#include <hip/hip_runtime.h>
#include <hip/hip_bf16.h>

// ---------------- problem constants ----------------
#define Bb 8
#define Tt 16
#define Nn 1024
#define DENC 768
#define Ss 7
#define Dd 192
#define HEADS 4
#define HD 48
#define ITERS 3
#define ROWS_ALL (Bb*Tt*Nn)          // 131072
#define SLOT_ELEMS (Bb*Ss*Dd)        // 10752
#define OUT_HALF (Bb*Tt*Ss*Dd)       // 172032

#define SCALE_F 0.07216878364870322f   // 192^-0.5
#define INV_SQRT_HD 0.14433756729740643f // 48^-0.5
#define EPS_F 1e-8f
#define LN_EPS 1e-5f

typedef __attribute__((ext_vector_type(16))) __bf16 v16bf;
typedef __attribute__((ext_vector_type(8)))  __bf16 v8bf;
typedef __attribute__((ext_vector_type(8)))  float  v8f;

// ============================================================
// 1) Pack Wk (scaled) | Wv into fragment-ready bf16 layout:
//    Bprep[kt][c][kk]  kt=0..23 (K tiles of 32), c=0..383, kk=0..31
// ============================================================
__global__ __launch_bounds__(256) void wprep_kernel(const float* __restrict__ Wk,
                                                    const float* __restrict__ Wv,
                                                    __bf16* __restrict__ Bp) {
  int idx = blockIdx.x * 256 + threadIdx.x;      // 24*384*32 = 294912
  if (idx >= 24 * 384 * 32) return;
  int kk = idx & 31;
  int c  = (idx >> 5) % 384;
  int kt = idx / (384 * 32);
  int j  = kt * 32 + kk;                          // 0..767
  float v = (c < Dd) ? Wk[j * Dd + c] * SCALE_F : Wv[j * Dd + (c - Dd)];
  Bp[idx] = (__bf16)v;
}

// ============================================================
// 2) LayerNorm of inputs over DENC, output bf16. One wave per row.
// ============================================================
__global__ __launch_bounds__(256) void ln_in_kernel(const float* __restrict__ x,
                                                    const float* __restrict__ g,
                                                    const float* __restrict__ be,
                                                    __bf16* __restrict__ out) {
  int wave = (blockIdx.x * 256 + threadIdx.x) >> 5;   // 0..131071
  int lane = threadIdx.x & 31;
  const float* row = x + (size_t)wave * DENC;
  float s = 0.f, s2 = 0.f, vals[24];
#pragma unroll
  for (int i = 0; i < 24; ++i) {
    float v = row[lane + i * 32];
    vals[i] = v; s += v; s2 += v * v;
  }
#pragma unroll
  for (int off = 16; off; off >>= 1) { s += __shfl_xor(s, off, 32); s2 += __shfl_xor(s2, off, 32); }
  float mean = s * (1.f / DENC);
  float var  = s2 * (1.f / DENC) - mean * mean;
  float rinv = rsqrtf(var + LN_EPS);
  __bf16* orow = out + (size_t)wave * DENC;
#pragma unroll
  for (int i = 0; i < 24; ++i) {
    int j = lane + i * 32;
    orow[j] = (__bf16)((vals[i] - mean) * rinv * g[j] + be[j]);
  }
}

// ============================================================
// 3) WMMA GEMM: [131072 x 768] bf16  @  [768 x 384] bf16 -> K|V f32
//    One wave = 16 rows x 64 cols (4 wmma tiles), 24 k-steps of 32.
// ============================================================
__global__ __launch_bounds__(256) void kv_gemm_kernel(const __bf16* __restrict__ X,
                                                      const __bf16* __restrict__ Bp,
                                                      float* __restrict__ kout,
                                                      float* __restrict__ vout) {
  int wave = (blockIdx.x * 256 + threadIdx.x) >> 5;   // 0..49151
  int lane = threadIdx.x & 31;
  int rowTile = wave / 6;                             // 0..8191
  int cg      = wave - rowTile * 6;                   // 0..5  (64 cols each)
  int rowBase = rowTile * 16;
  int colBase = cg * 64;

  int mrow  = rowBase + (lane & 15);
  int koffA = (lane < 16) ? 0 : 16;     // A: contiguous k = 16h + s (matches B)
  int koffB = (lane < 16) ? 0 : 16;     // B: contiguous k = 16h + s (matches A)

  v8f acc0 = {}, acc1 = {}, acc2 = {}, acc3 = {};
  const __bf16* arow = X + (size_t)mrow * DENC;

  for (int kt = 0; kt < 24; ++kt) {
    int kk = kt * 32;
    v8bf alo = *(const v8bf*)(arow + kk + koffA);
    v8bf ahi = *(const v8bf*)(arow + kk + 8 + koffA);
    v16bf a;
#pragma unroll
    for (int i = 0; i < 8; ++i) { a[i] = alo[i]; a[8 + i] = ahi[i]; }

    const __bf16* bbase = Bp + ((size_t)kt * 384 + colBase + (lane & 15)) * 32 + koffB;
    v16bf b0 = *(const v16bf*)(bbase + 0 * 16 * 32);
    v16bf b1 = *(const v16bf*)(bbase + 1 * 16 * 32);
    v16bf b2 = *(const v16bf*)(bbase + 2 * 16 * 32);
    v16bf b3 = *(const v16bf*)(bbase + 3 * 16 * 32);

    acc0 = __builtin_amdgcn_wmma_f32_16x16x32_bf16(false, a, false, b0, (short)0, acc0, false, false);
    acc1 = __builtin_amdgcn_wmma_f32_16x16x32_bf16(false, a, false, b1, (short)0, acc1, false, false);
    acc2 = __builtin_amdgcn_wmma_f32_16x16x32_bf16(false, a, false, b2, (short)0, acc2, false, false);
    acc3 = __builtin_amdgcn_wmma_f32_16x16x32_bf16(false, a, false, b3, (short)0, acc3, false, false);
  }

  // D layout: element r of acc -> row rowBase + r + 8*(lane>=16), col tile n = lane&15
  int mstore = rowBase + ((lane >> 4) * 8);
  int nlane  = lane & 15;
  v8f accs[4] = {acc0, acc1, acc2, acc3};
#pragma unroll
  for (int bt = 0; bt < 4; ++bt) {
    int n = colBase + bt * 16 + nlane;
#pragma unroll
    for (int r = 0; r < 8; ++r) {
      int m = mstore + r;
      float val = accs[bt][r];
      if (n < Dd) kout[(size_t)m * Dd + n] = val;
      else        vout[(size_t)m * Dd + (n - Dd)] = val;
    }
  }
}

// ============================================================
// 4) slots0 = mu + exp(log_sigma) * noise
// ============================================================
__global__ __launch_bounds__(256) void slots0_kernel(const float* __restrict__ noise,
                                                     const float* __restrict__ mu,
                                                     const float* __restrict__ lsig,
                                                     float* __restrict__ cur) {
  int idx = blockIdx.x * 256 + threadIdx.x;
  if (idx >= SLOT_ELEMS) return;
  int d = idx % Dd;
  cur[idx] = mu[d] + __expf(lsig[d]) * noise[idx];
}

// ============================================================
// 5) copy slots -> d_out[b][t][s][d]
// ============================================================
__global__ __launch_bounds__(256) void copy_slots_kernel(const float* __restrict__ src,
                                                         float* __restrict__ dst, int t) {
  int idx = blockIdx.x * 256 + threadIdx.x;
  if (idx >= SLOT_ELEMS) return;
  int b = idx / (Ss * Dd), rem = idx % (Ss * Dd);
  dst[((size_t)b * Tt + t) * (Ss * Dd) + rem] = src[idx];
}

// ============================================================
// 6) q = LN(cur) @ Wq   — block per (b,s), 192 threads
// ============================================================
__global__ __launch_bounds__(192) void slot_q_kernel(const float* __restrict__ cur,
                                                     const float* __restrict__ g,
                                                     const float* __restrict__ be,
                                                     const float* __restrict__ Wq,
                                                     float* __restrict__ q) {
  __shared__ float hln[Dd];
  __shared__ float red[16];
  __shared__ float stat[2];
  int row = blockIdx.x;          // 0..55
  int d = threadIdx.x;
  float v = cur[row * Dd + d];
  float s = v, s2 = v * v;
#pragma unroll
  for (int off = 16; off; off >>= 1) { s += __shfl_xor(s, off, 32); s2 += __shfl_xor(s2, off, 32); }
  int w = d >> 5, lane = d & 31;
  if (lane == 0) { red[w] = s; red[8 + w] = s2; }
  __syncthreads();
  if (d == 0) {
    float a = 0.f, b2 = 0.f;
    for (int i = 0; i < 6; ++i) { a += red[i]; b2 += red[8 + i]; }
    float mean = a * (1.f / Dd);
    float var  = b2 * (1.f / Dd) - mean * mean;
    stat[0] = mean; stat[1] = rsqrtf(var + LN_EPS);
  }
  __syncthreads();
  hln[d] = (v - stat[0]) * stat[1] * g[d] + be[d];
  __syncthreads();
  float acc = 0.f;
  for (int j = 0; j < Dd; ++j) acc += hln[j] * Wq[j * Dd + d];
  q[row * Dd + d] = acc;
}

// ============================================================
// 7) dots + softmax over slots.  One wave per (b,n).
// ============================================================
__global__ __launch_bounds__(256) void slot_attn_kernel(const float* __restrict__ kbuf,
                                                        const float* __restrict__ q,
                                                        float* __restrict__ attn, int t) {
  int wave = (blockIdx.x * 256 + threadIdx.x) >> 5;   // 0..8191
  int lane = threadIdx.x & 31;
  int b = wave >> 10;
  int n = wave & (Nn - 1);
  const float* krow = kbuf + (((size_t)(b * Tt + t)) * Nn + n) * Dd;
  float p0=0,p1=0,p2=0,p3=0,p4=0,p5=0,p6=0;
  for (int j = lane; j < Dd; j += 32) {
    float kv = krow[j];
    const float* qb = q + b * Ss * Dd + j;
    p0 += kv * qb[0*Dd]; p1 += kv * qb[1*Dd]; p2 += kv * qb[2*Dd];
    p3 += kv * qb[3*Dd]; p4 += kv * qb[4*Dd]; p5 += kv * qb[5*Dd];
    p6 += kv * qb[6*Dd];
  }
#pragma unroll
  for (int off = 16; off; off >>= 1) {
    p0 += __shfl_xor(p0, off, 32); p1 += __shfl_xor(p1, off, 32);
    p2 += __shfl_xor(p2, off, 32); p3 += __shfl_xor(p3, off, 32);
    p4 += __shfl_xor(p4, off, 32); p5 += __shfl_xor(p5, off, 32);
    p6 += __shfl_xor(p6, off, 32);
  }
  float m = fmaxf(fmaxf(fmaxf(p0,p1),fmaxf(p2,p3)),fmaxf(fmaxf(p4,p5),p6));
  float e0=__expf(p0-m), e1=__expf(p1-m), e2=__expf(p2-m), e3=__expf(p3-m);
  float e4=__expf(p4-m), e5=__expf(p5-m), e6=__expf(p6-m);
  float rt = 1.f / (e0+e1+e2+e3+e4+e5+e6);
  if (lane < Ss) {
    float e = (lane==0)?e0:(lane==1)?e1:(lane==2)?e2:(lane==3)?e3:(lane==4)?e4:(lane==5)?e5:e6;
    attn[((size_t)b * Nn + n) * Ss + lane] = e * rt + EPS_F;
  }
}

// ============================================================
// 8) upd[b,s,d] = sum_n attn*v / sum_n attn.  Block per b, 192 threads.
//    Denominator computed redundantly per thread (deterministic, no atomics).
// ============================================================
__global__ __launch_bounds__(192) void slot_upd_kernel(const float* __restrict__ vbuf,
                                                       const float* __restrict__ attn,
                                                       float* __restrict__ upd, int t) {
  int b = blockIdx.x;
  int d = threadIdx.x;
  float a0=0,a1=0,a2=0,a3=0,a4=0,a5=0,a6=0;
  float c0=0,c1=0,c2=0,c3=0,c4=0,c5=0,c6=0;
  const float* vbase = vbuf + ((size_t)(b * Tt + t)) * Nn * Dd;
  const float* abase = attn + (size_t)b * Nn * Ss;
  for (int n = 0; n < Nn; ++n) {
    float vv = vbase[n * Dd + d];
    const float* ar = abase + n * Ss;
    float w0=ar[0],w1=ar[1],w2=ar[2],w3=ar[3],w4=ar[4],w5=ar[5],w6=ar[6];
    a0 += w0*vv; a1 += w1*vv; a2 += w2*vv; a3 += w3*vv; a4 += w4*vv; a5 += w5*vv; a6 += w6*vv;
    c0 += w0; c1 += w1; c2 += w2; c3 += w3; c4 += w4; c5 += w5; c6 += w6;
  }
  float* ub = upd + b * Ss * Dd + d;
  ub[0*Dd]=a0/c0; ub[1*Dd]=a1/c1; ub[2*Dd]=a2/c2; ub[3*Dd]=a3/c3;
  ub[4*Dd]=a4/c4; ub[5*Dd]=a5/c5; ub[6*Dd]=a6/c6;
}

// ============================================================
// 9) GRU cell.  Block per (b,s), 192 threads.
// ============================================================
__global__ __launch_bounds__(192) void gru_kernel(const float* __restrict__ upd,
                                                  const float* __restrict__ cur,
                                                  const float* __restrict__ Wih,
                                                  const float* __restrict__ Whh,
                                                  const float* __restrict__ bih,
                                                  const float* __restrict__ bhh,
                                                  float* __restrict__ dst) {
  __shared__ float xs[Dd], hs[Dd];
  int row = blockIdx.x, d = threadIdx.x;
  xs[d] = upd[row * Dd + d];
  hs[d] = cur[row * Dd + d];
  __syncthreads();
  float gi[3], gh[3];
#pragma unroll
  for (int gdx = 0; gdx < 3; ++gdx) {
    const float* wi = Wih + (size_t)(gdx * Dd + d) * Dd;
    const float* wh = Whh + (size_t)(gdx * Dd + d) * Dd;
    float a = bih[gdx * Dd + d], c = bhh[gdx * Dd + d];
    for (int j = 0; j < Dd; ++j) { a += xs[j] * wi[j]; c += hs[j] * wh[j]; }
    gi[gdx] = a; gh[gdx] = c;
  }
  float r = 1.f / (1.f + __expf(-(gi[0] + gh[0])));
  float z = 1.f / (1.f + __expf(-(gi[1] + gh[1])));
  float nn = tanhf(gi[2] + r * gh[2]);
  dst[row * Dd + d] = (1.f - z) * nn + z * hs[d];
}

// ============================================================
// 10) slot MLP residual (in-place).  Block per (b,s), 256 threads.
// ============================================================
__global__ __launch_bounds__(256) void slot_mlp_kernel(float* __restrict__ s,
                                                       const float* __restrict__ g,
                                                       const float* __restrict__ be,
                                                       const float* __restrict__ W1,
                                                       const float* __restrict__ b1,
                                                       const float* __restrict__ W2,
                                                       const float* __restrict__ b2) {
  __shared__ float ss[Dd], hln[Dd], hid[4 * Dd];
  __shared__ float red[16], stat[2];
  int row = blockIdx.x, tid = threadIdx.x;
  float v = 0.f;
  if (tid < Dd) { v = s[row * Dd + tid]; ss[tid] = v; }
  float su = (tid < Dd) ? v : 0.f, su2 = (tid < Dd) ? v * v : 0.f;
#pragma unroll
  for (int off = 16; off; off >>= 1) { su += __shfl_xor(su, off, 32); su2 += __shfl_xor(su2, off, 32); }
  int w = tid >> 5, lane = tid & 31;
  if (lane == 0) { red[w] = su; red[8 + w] = su2; }
  __syncthreads();
  if (tid == 0) {
    float a = 0.f, b = 0.f;
    for (int i = 0; i < 8; ++i) { a += red[i]; b += red[8 + i]; }
    float mean = a * (1.f / Dd);
    float var  = b * (1.f / Dd) - mean * mean;
    stat[0] = mean; stat[1] = rsqrtf(var + LN_EPS);
  }
  __syncthreads();
  if (tid < Dd) hln[tid] = (ss[tid] - stat[0]) * stat[1] * g[tid] + be[tid];
  __syncthreads();
  for (int o = tid; o < 4 * Dd; o += 256) {
    float a = b1[o];
    for (int j = 0; j < Dd; ++j) a += hln[j] * W1[j * 4 * Dd + o];
    hid[o] = fmaxf(a, 0.f);
  }
  __syncthreads();
  if (tid < Dd) {
    float a = b2[tid];
    for (int o = 0; o < 4 * Dd; ++o) a += hid[o] * W2[o * Dd + tid];
    s[row * Dd + tid] = ss[tid] + a;
  }
}

// ============================================================
// 11) Predictor: one fused pre-LN transformer block.  Block per batch b.
// ============================================================
__global__ __launch_bounds__(256) void predictor_kernel(const float* __restrict__ cur,
    float* __restrict__ nxt,
    const float* __restrict__ ln1g, const float* __restrict__ ln1b,
    const float* __restrict__ Wq, const float* __restrict__ bq,
    const float* __restrict__ Wk, const float* __restrict__ bk,
    const float* __restrict__ Wv, const float* __restrict__ bv,
    const float* __restrict__ Wo, const float* __restrict__ bo,
    const float* __restrict__ ln2g, const float* __restrict__ ln2b,
    const float* __restrict__ W1, const float* __restrict__ b1,
    const float* __restrict__ W2, const float* __restrict__ b2) {
  __shared__ float ss[Ss * Dd], hh[Ss * Dd], qq[Ss * Dd], kk2[Ss * Dd], vv2[Ss * Dd];
  __shared__ float hid[Ss * 4 * Dd];
  __shared__ float sc[HEADS * Ss * Ss];
  int b = blockIdx.x, tid = threadIdx.x;
  int w = tid >> 5, lane = tid & 31;

  for (int i = tid; i < Ss * Dd; i += 256) ss[i] = cur[b * Ss * Dd + i];
  __syncthreads();
  // LN1 -> hh   (wave w handles row w)
  if (w < Ss) {
    float s = 0.f, s2 = 0.f, vals[6];
#pragma unroll
    for (int i = 0; i < 6; ++i) { float x = ss[w * Dd + lane + i * 32]; vals[i] = x; s += x; s2 += x * x; }
#pragma unroll
    for (int off = 16; off; off >>= 1) { s += __shfl_xor(s, off, 32); s2 += __shfl_xor(s2, off, 32); }
    float mean = s * (1.f / Dd), var = s2 * (1.f / Dd) - mean * mean, rinv = rsqrtf(var + LN_EPS);
#pragma unroll
    for (int i = 0; i < 6; ++i) { int j = lane + i * 32; hh[w * Dd + j] = (vals[i] - mean) * rinv * ln1g[j] + ln1b[j]; }
  }
  __syncthreads();
  // q,k,v projections
  for (int idx = tid; idx < 3 * Ss * Dd; idx += 256) {
    int which = idx / (Ss * Dd), rem = idx - which * Ss * Dd;
    int row = rem / Dd, d = rem - row * Dd;
    const float* W  = (which == 0) ? Wq : (which == 1) ? Wk : Wv;
    const float* bb = (which == 0) ? bq : (which == 1) ? bk : bv;
    float a = bb[d];
    for (int j = 0; j < Dd; ++j) a += hh[row * Dd + j] * W[j * Dd + d];
    float* dstp = (which == 0) ? qq : (which == 1) ? kk2 : vv2;
    dstp[rem] = a;
  }
  __syncthreads();
  // attention scores
  for (int idx = tid; idx < HEADS * Ss * Ss; idx += 256) {
    int h = idx / (Ss * Ss), rr = (idx - h * Ss * Ss) / Ss, c = idx % Ss;
    float a = 0.f;
    for (int j = 0; j < HD; ++j) a += qq[rr * Dd + h * HD + j] * kk2[c * Dd + h * HD + j];
    sc[idx] = a * INV_SQRT_HD;
  }
  __syncthreads();
  if (tid < HEADS * Ss) {
    float* p = sc + tid * Ss;
    float m = p[0];
    for (int c = 1; c < Ss; ++c) m = fmaxf(m, p[c]);
    float tot = 0.f;
    for (int c = 0; c < Ss; ++c) { p[c] = __expf(p[c] - m); tot += p[c]; }
    float rt = 1.f / tot;
    for (int c = 0; c < Ss; ++c) p[c] *= rt;
  }
  __syncthreads();
  // attention output -> reuse hh
  for (int idx = tid; idx < Ss * Dd; idx += 256) {
    int row = idx / Dd, d = idx - row * Dd, h = d / HD;
    float a = 0.f;
    for (int c = 0; c < Ss; ++c) a += sc[(h * Ss + row) * Ss + c] * vv2[c * Dd + d];
    hh[idx] = a;
  }
  __syncthreads();
  // residual: ss += hh @ Wo + bo
  for (int idx = tid; idx < Ss * Dd; idx += 256) {
    int row = idx / Dd, d = idx - row * Dd;
    float a = bo[d];
    for (int j = 0; j < Dd; ++j) a += hh[row * Dd + j] * Wo[j * Dd + d];
    ss[idx] += a;
  }
  __syncthreads();
  // LN2 -> reuse qq
  if (w < Ss) {
    float s = 0.f, s2 = 0.f, vals[6];
#pragma unroll
    for (int i = 0; i < 6; ++i) { float x = ss[w * Dd + lane + i * 32]; vals[i] = x; s += x; s2 += x * x; }
#pragma unroll
    for (int off = 16; off; off >>= 1) { s += __shfl_xor(s, off, 32); s2 += __shfl_xor(s2, off, 32); }
    float mean = s * (1.f / Dd), var = s2 * (1.f / Dd) - mean * mean, rinv = rsqrtf(var + LN_EPS);
#pragma unroll
    for (int i = 0; i < 6; ++i) { int j = lane + i * 32; qq[w * Dd + j] = (vals[i] - mean) * rinv * ln2g[j] + ln2b[j]; }
  }
  __syncthreads();
  // FFN hidden
  for (int idx = tid; idx < Ss * 4 * Dd; idx += 256) {
    int row = idx / (4 * Dd), o = idx - row * 4 * Dd;
    float a = b1[o];
    for (int j = 0; j < Dd; ++j) a += qq[row * Dd + j] * W1[j * 4 * Dd + o];
    hid[idx] = fmaxf(a, 0.f);
  }
  __syncthreads();
  // FFN out + residual
  for (int idx = tid; idx < Ss * Dd; idx += 256) {
    int row = idx / Dd, d = idx - row * Dd;
    float a = b2[d];
    for (int o = 0; o < 4 * Dd; ++o) a += hid[row * 4 * Dd + o] * W2[o * Dd + d];
    nxt[b * Ss * Dd + idx] = ss[idx] + a;
  }
}

// ============================================================
// host launcher
// ============================================================
extern "C" void kernel_launch(void* const* d_in, const int* in_sizes, int n_in,
                              void* d_out, int out_size, void* d_ws, size_t ws_size,
                              hipStream_t stream) {
  const float* inputs   = (const float*)d_in[0];
  const float* noise    = (const float*)d_in[1];
  const float* slots_mu = (const float*)d_in[2];
  const float* slots_ls = (const float*)d_in[3];
  const float* Wq       = (const float*)d_in[4];
  const float* Wk       = (const float*)d_in[5];
  const float* Wv       = (const float*)d_in[6];
  const float* gWih     = (const float*)d_in[7];
  const float* gWhh     = (const float*)d_in[8];
  const float* gbih     = (const float*)d_in[9];
  const float* gbhh     = (const float*)d_in[10];
  const float* mW1      = (const float*)d_in[11];
  const float* mb1      = (const float*)d_in[12];
  const float* mW2      = (const float*)d_in[13];
  const float* mb2      = (const float*)d_in[14];
  const float* ln_in_g  = (const float*)d_in[15];
  const float* ln_in_b  = (const float*)d_in[16];
  const float* ln_sl_g  = (const float*)d_in[17];
  const float* ln_sl_b  = (const float*)d_in[18];
  const float* ln_ff_g  = (const float*)d_in[19];
  const float* ln_ff_b  = (const float*)d_in[20];
  const float* p_ln1_g  = (const float*)d_in[21];
  const float* p_ln1_b  = (const float*)d_in[22];
  const float* p_Wq     = (const float*)d_in[23];
  const float* p_bq     = (const float*)d_in[24];
  const float* p_Wk     = (const float*)d_in[25];
  const float* p_bk     = (const float*)d_in[26];
  const float* p_Wv     = (const float*)d_in[27];
  const float* p_bv     = (const float*)d_in[28];
  const float* p_Wo     = (const float*)d_in[29];
  const float* p_bo     = (const float*)d_in[30];
  const float* p_ln2_g  = (const float*)d_in[31];
  const float* p_ln2_b  = (const float*)d_in[32];
  const float* p_W1     = (const float*)d_in[33];
  const float* p_b1     = (const float*)d_in[34];
  const float* p_W2     = (const float*)d_in[35];
  const float* p_b2     = (const float*)d_in[36];

  // workspace layout
  char* ws = (char*)d_ws;
  size_t off = 0;
  auto carve = [&](size_t bytes) { void* p = ws + off; off = (off + bytes + 255) & ~(size_t)255; return p; };
  __bf16* xln   = (__bf16*)carve((size_t)ROWS_ALL * DENC * sizeof(__bf16));   // 201 MB
  __bf16* bprep = (__bf16*)carve((size_t)24 * 384 * 32 * sizeof(__bf16));
  float*  kbuf  = (float*)carve((size_t)ROWS_ALL * Dd * sizeof(float));       // 100 MB
  float*  vbuf  = (float*)carve((size_t)ROWS_ALL * Dd * sizeof(float));       // 100 MB
  float*  slotsA= (float*)carve((size_t)SLOT_ELEMS * sizeof(float));
  float*  slotsB= (float*)carve((size_t)SLOT_ELEMS * sizeof(float));
  float*  qbuf  = (float*)carve((size_t)SLOT_ELEMS * sizeof(float));
  float*  attnb = (float*)carve((size_t)Bb * Nn * Ss * sizeof(float));
  float*  updb  = (float*)carve((size_t)SLOT_ELEMS * sizeof(float));
  (void)ws_size; (void)in_sizes; (void)n_in; (void)out_size;

  float* out_list = (float*)d_out;            // slots_list  [B,T,S,D]
  float* out_init = out_list + OUT_HALF;      // slots_init  [B,T,S,D]

  // heavy phase
  wprep_kernel<<<(24 * 384 * 32) / 256, 256, 0, stream>>>(Wk, Wv, bprep);
  ln_in_kernel<<<ROWS_ALL / 8, 256, 0, stream>>>(inputs, ln_in_g, ln_in_b, xln);
  kv_gemm_kernel<<<(ROWS_ALL / 16) * 6 / 8, 256, 0, stream>>>(xln, bprep, kbuf, vbuf);
  slots0_kernel<<<SLOT_ELEMS / 256, 256, 0, stream>>>(noise, slots_mu, slots_ls, slotsA);

  float* cur = slotsA;
  float* alt = slotsB;
  for (int t = 0; t < Tt; ++t) {
    copy_slots_kernel<<<SLOT_ELEMS / 256, 256, 0, stream>>>(cur, out_init, t);
    for (int it = 0; it < ITERS; ++it) {
      slot_q_kernel<<<Bb * Ss, Dd, 0, stream>>>(cur, ln_sl_g, ln_sl_b, Wq, qbuf);
      slot_attn_kernel<<<(Bb * Nn) / 8, 256, 0, stream>>>(kbuf, qbuf, attnb, t);
      slot_upd_kernel<<<Bb, Dd, 0, stream>>>(vbuf, attnb, updb, t);
      gru_kernel<<<Bb * Ss, Dd, 0, stream>>>(updb, cur, gWih, gWhh, gbih, gbhh, alt);
      { float* tmp = cur; cur = alt; alt = tmp; }
      if (it < ITERS - 1)
        slot_mlp_kernel<<<Bb * Ss, 256, 0, stream>>>(cur, ln_ff_g, ln_ff_b, mW1, mb1, mW2, mb2);
    }
    copy_slots_kernel<<<SLOT_ELEMS / 256, 256, 0, stream>>>(cur, out_list, t);
    predictor_kernel<<<Bb, 256, 0, stream>>>(cur, alt,
        p_ln1_g, p_ln1_b, p_Wq, p_bq, p_Wk, p_bk, p_Wv, p_bv,
        p_Wo, p_bo, p_ln2_g, p_ln2_b, p_W1, p_b1, p_W2, p_b2);
    { float* tmp = cur; cur = alt; alt = tmp; }
  }
}
